// LightningIndexer_40845138985325
// MI455X (gfx1250) — 16-bit and 8-bit matrix operand order repaired and checked by reading, compiled, not run on hardware
//
#include <hip/hip_runtime.h>
#include <hip/hip_bf16.h>

#define B_  1
#define T_  2048
#define D_  2048
#define H_  16
#define L_  512
#define RD_ 64
#define HD_ 128
#define QK_ 192

typedef __attribute__((ext_vector_type(16))) int    v16i;
typedef __attribute__((ext_vector_type(8)))  int    v8i;
typedef __attribute__((ext_vector_type(8)))  float  v8f;
typedef __attribute__((ext_vector_type(16))) __bf16 v16bf;

typedef int tv4 __attribute__((vector_size(4 * sizeof(int))));
using tv4_g = __attribute__((address_space(1))) tv4;
using tv4_l = __attribute__((address_space(3))) tv4;

#ifndef __has_builtin
#define __has_builtin(x) 0
#endif
#if __has_builtin(__builtin_amdgcn_global_load_async_to_lds_b128)
#define USE_ASYNC_LDS 1
#else
#define USE_ASYNC_LDS 0
#endif

__device__ inline void cp16_g2l(const void* g, void* l) {
#if USE_ASYNC_LDS
    __builtin_amdgcn_global_load_async_to_lds_b128((tv4_g*)g, (tv4_l*)l, 0, 0);
#else
    *(int4*)l = *(const int4*)g;
#endif
}
__device__ inline void wait_async0() {
#if USE_ASYNC_LDS
    asm volatile("s_wait_asynccnt 0" ::: "memory");
#endif
}

__device__ inline unsigned char f32_to_e4m3(float f) {
    unsigned int u  = __float_as_uint(f);
    unsigned char s = (unsigned char)((u >> 24) & 0x80u);
    unsigned int au = u & 0x7fffffffu;
    if (au >= 0x7f800000u) return (unsigned char)(s | 0x7fu);
    float a = __uint_as_float(au);
    if (a < 0.015625f) {
        int q = (int)rintf(a * 512.0f);
        return (unsigned char)(s | (unsigned char)q);
    }
    unsigned int lsb = (au >> 20) & 1u;
    unsigned int r   = (au + 0x0007FFFFu + lsb) & 0xFFF00000u;
    float ra = __uint_as_float(r);
    if (ra > 448.0f) return (unsigned char)(s | 0x7Eu);
    int e = (int)(r >> 23) - 127;
    unsigned int m = (r >> 20) & 7u;
    return (unsigned char)(s | (unsigned char)(((e + 7) << 3) | m));
}

__device__ inline unsigned short f32_to_bf16_rne(float f) {
    unsigned int u = __float_as_uint(f);
    unsigned int lsb = (u >> 16) & 1u;
    u += 0x7FFFu + lsb;
    return (unsigned short)(u >> 16);
}
__device__ inline float bf16_to_f32(unsigned short h) {
    return __uint_as_float(((unsigned int)h) << 16);
}

__global__ __launch_bounds__(256) void split_bf16(const float* __restrict__ in,
                                                  unsigned short* __restrict__ hi,
                                                  unsigned short* __restrict__ lo,
                                                  int n) {
    int i = blockIdx.x * 256 + threadIdx.x;
    if (i >= n) return;
    float f = in[i];
    unsigned short h = f32_to_bf16_rne(f);
    unsigned short l = f32_to_bf16_rne(f - bf16_to_f32(h));
    hi[i] = h;
    lo[i] = l;
}

__global__ __launch_bounds__(256) void split_bf16_T(const float* __restrict__ in,
                                                    unsigned short* __restrict__ hiT,
                                                    unsigned short* __restrict__ loT,
                                                    int K, int N) {
    int idx = blockIdx.x * 256 + threadIdx.x;
    if (idx >= K * N) return;
    int k = idx / N, n = idx % N;
    float f = in[idx];
    unsigned short h = f32_to_bf16_rne(f);
    unsigned short l = f32_to_bf16_rne(f - bf16_to_f32(h));
    hiT[(long)n * K + k] = h;
    loT[(long)n * K + k] = l;
}

template <int NT>
__global__ __launch_bounds__(256) void gemm_bf16x3(const unsigned short* __restrict__ Ahi,
                                                   const unsigned short* __restrict__ Alo,
                                                   const unsigned short* __restrict__ BhiT,
                                                   const unsigned short* __restrict__ BloT,
                                                   float* __restrict__ C,
                                                   int M, int N, int K) {
    const int lane = threadIdx.x & 31;
    const int wv   = threadIdx.x >> 5;
    const int mt   = blockIdx.y * 8 + wv;
    const int n0   = blockIdx.x * (NT * 16);
    const int r0   = mt * 16;
    const int l15   = lane & 15;
    const bool hiL  = lane >= 16;
    const int aoffB = hiL ? 16 : 0;
    const int boffB = hiL ? 16 : 0;

    v8f acc[NT];
    #pragma unroll
    for (int j = 0; j < NT; ++j) acc[j] = (v8f){0.f,0.f,0.f,0.f,0.f,0.f,0.f,0.f};

    const char* arow_hi = (const char*)(Ahi + (long)(r0 + l15) * K);
    const char* arow_lo = (const char*)(Alo + (long)(r0 + l15) * K);
    const char* bh[NT];
    const char* bl[NT];
    #pragma unroll
    for (int j = 0; j < NT; ++j) {
        bh[j] = (const char*)(BhiT + (long)(n0 + j * 16 + l15) * K);
        bl[j] = (const char*)(BloT + (long)(n0 + j * 16 + l15) * K);
    }

    for (int k0 = 0; k0 < K; k0 += 32) {
        const long kb = (long)k0 * 2;
        int4 u0 = *(const int4*)(arow_hi + kb + aoffB);
        int4 u1 = *(const int4*)(arow_hi + kb + 32 + aoffB);
        int4 u2 = *(const int4*)(arow_lo + kb + aoffB);
        int4 u3 = *(const int4*)(arow_lo + kb + 32 + aoffB);
        v8i ta = {u0.x, u0.y, u0.z, u0.w, u1.x, u1.y, u1.z, u1.w};
        v8i tl2 = {u2.x, u2.y, u2.z, u2.w, u3.x, u3.y, u3.z, u3.w};
        v16bf ahi = __builtin_bit_cast(v16bf, ta);
        v16bf alo = __builtin_bit_cast(v16bf, tl2);

        #pragma unroll
        for (int j = 0; j < NT; ++j) {
            int4 w0 = *(const int4*)(bh[j] + kb + boffB);
            int4 w1 = *(const int4*)(bh[j] + kb + boffB + 32);
            int4 w2 = *(const int4*)(bl[j] + kb + boffB);
            int4 w3 = *(const int4*)(bl[j] + kb + boffB + 32);
            v8i th = {w0.x, w0.y, w0.z, w0.w, w1.x, w1.y, w1.z, w1.w};
            v8i tll = {w2.x, w2.y, w2.z, w2.w, w3.x, w3.y, w3.z, w3.w};
            v16bf bhi = __builtin_bit_cast(v16bf, th);
            v16bf blo = __builtin_bit_cast(v16bf, tll);
            acc[j] = __builtin_amdgcn_wmma_f32_16x16x32_bf16(false, ahi, false, bhi,
                                                             (short)0, acc[j], false, false);
            acc[j] = __builtin_amdgcn_wmma_f32_16x16x32_bf16(false, ahi, false, blo,
                                                             (short)0, acc[j], false, false);
            acc[j] = __builtin_amdgcn_wmma_f32_16x16x32_bf16(false, alo, false, bhi,
                                                             (short)0, acc[j], false, false);
        }
    }

    const int rowoff = hiL ? 8 : 0;
    #pragma unroll
    for (int j = 0; j < NT; ++j) {
        #pragma unroll
        for (int v = 0; v < 8; ++v) {
            int r = r0 + v + rowoff;
            int c = n0 + j * 16 + l15;
            C[(long)r * N + c] = acc[j][v];
        }
    }
}

__global__ __launch_bounds__(256) void hw_sigmoid_t(const float* __restrict__ hwlin,
                                                    float* __restrict__ hwT) {
    int idx = blockIdx.x * 256 + threadIdx.x;
    if (idx >= T_ * H_) return;
    int t = idx >> 4, h = idx & 15;
    float v = hwlin[idx];
    hwT[h * T_ + t] = 1.0f / (1.0f + __expf(-v));
}

__global__ __launch_bounds__(256) void fwht128_quant(const float* __restrict__ in,
                                                     int inRowStride,
                                                     unsigned char* __restrict__ out,
                                                     long outHeadStride, int nH,
                                                     float scale) {
    int gw   = (blockIdx.x * blockDim.x + threadIdx.x) >> 5;
    int lane = threadIdx.x & 31;
    int t = gw / nH, h = gw % nH;
    if (t >= T_) return;
    const float4* p = (const float4*)(in + (long)t * inRowStride + h * 128);
    float4 vv = p[lane];
    float v[4] = {vv.x, vv.y, vv.z, vv.w};
    { float a = v[0], b = v[1]; v[0] = a + b; v[1] = a - b;
      a = v[2]; b = v[3];       v[2] = a + b; v[3] = a - b; }
    { float a = v[0], b = v[2]; v[0] = a + b; v[2] = a - b;
      a = v[1]; b = v[3];       v[1] = a + b; v[3] = a - b; }
    #pragma unroll
    for (int s = 1; s <= 16; s <<= 1) {
        bool up = (lane & s) != 0;
        #pragma unroll
        for (int j = 0; j < 4; ++j) {
            float pr = __shfl_xor(v[j], s, 32);
            v[j] = up ? (pr - v[j]) : (v[j] + pr);
        }
    }
    unsigned int pk = 0;
    #pragma unroll
    for (int j = 0; j < 4; ++j)
        pk |= ((unsigned int)f32_to_e4m3(v[j] * scale)) << (8 * j);
    *(unsigned int*)(out + (long)h * outHeadStride + (long)t * QK_ + lane * 4) = pk;
}

__global__ __launch_bounds__(256) void rope_fwht64_quant(const float* __restrict__ in,
                                                         int inRowStride,
                                                         unsigned char* __restrict__ out,
                                                         long outHeadStride, int nH,
                                                         float scale) {
    int gw   = (blockIdx.x * blockDim.x + threadIdx.x) >> 5;
    int lane = threadIdx.x & 31;
    int t = gw / nH, h = gw % nH;
    if (t >= T_) return;
    const float2* p = (const float2*)(in + (long)t * inRowStride + h * 64);
    float2 vv = p[lane];
    float v[2] = {vv.x, vv.y};
    float prt[2];
    #pragma unroll
    for (int j = 0; j < 2; ++j) prt[j] = __shfl_xor(v[j], 16, 32);
    #pragma unroll
    for (int j = 0; j < 2; ++j) {
        int e = lane * 2 + j;
        int i = e & 31;
        float ang = (float)t * __powf(10000.0f, -(float)i / 32.0f);
        float c = __cosf(ang), sn = __sinf(ang);
        v[j] = (e < 32) ? (v[j] * c - prt[j] * sn) : (v[j] * c + prt[j] * sn);
    }
    { float a = v[0], b = v[1]; v[0] = a + b; v[1] = a - b; }
    #pragma unroll
    for (int s = 1; s <= 16; s <<= 1) {
        bool up = (lane & s) != 0;
        #pragma unroll
        for (int j = 0; j < 2; ++j) {
            float pr = __shfl_xor(v[j], s, 32);
            v[j] = up ? (pr - v[j]) : (v[j] + pr);
        }
    }
    unsigned short pk = (unsigned short)f32_to_e4m3(v[0] * scale)
                      | ((unsigned short)f32_to_e4m3(v[1] * scale) << 8);
    *(unsigned short*)(out + (long)h * outHeadStride + (long)t * QK_ + 128 + lane * 2) = pk;
}

__global__ __launch_bounds__(256) void attn_scores(const unsigned char* __restrict__ Qp,
                                                   const unsigned char* __restrict__ Kp,
                                                   const float* __restrict__ hwT,
                                                   float* __restrict__ out) {
    __shared__ __align__(16) unsigned char KpS[128 * QK_];
    __shared__ __align__(16) unsigned char QpS[H_ * 16 * QK_];
    const int tid  = threadIdx.x;
    const int lane = tid & 31;
    const int wv   = tid >> 5;
    const int rt   = blockIdx.y;
    const int ctg  = blockIdx.x;
    const int ct   = ctg * 8 + wv;
    const int r0 = rt * 16;
    const int n  = lane & 15;
    const bool hi = lane >= 16;
    const int rowoff = hi ? 8 : 0;

    const bool blockLive = (ctg * 8 <= rt);
    if (blockLive) {
        const unsigned char* kg = Kp + (long)ctg * 128 * QK_;
        #pragma unroll
        for (int i = 0; i < 6; ++i) {
            int o = tid * 16 + i * 4096;
            cp16_g2l(kg + o, KpS + o);
        }
        #pragma unroll
        for (int i = 0; i < 12; ++i) {
            int chunk = tid + i * 256;
            int h     = chunk / 192;
            int inner = chunk % 192;
            cp16_g2l(Qp + ((long)h * T_ + r0) * QK_ + inner * 16,
                     QpS + h * (16 * QK_) + inner * 16);
        }
        wait_async0();
        __syncthreads();
    }

    if (ct > rt) {
        #pragma unroll
        for (int v = 0; v < 8; ++v)
            out[(long)(r0 + v + rowoff) * T_ + (ct * 16 + n)] = 0.0f;
        return;
    }

    const unsigned char* kc = KpS + (wv * 16 + n) * QK_;
    const int boff = hi ? 16 : 0;
    int4 b0 = *(const int4*)(kc + boff + 0);
    int4 b1 = *(const int4*)(kc + boff + 32);
    int4 b2 = *(const int4*)(kc + boff + 64);
    int4 b3 = *(const int4*)(kc + boff + 96);
    v16i b128 = {b0.x, b0.y, b0.z, b0.w, b1.x, b1.y, b1.z, b1.w,
                 b2.x, b2.y, b2.z, b2.w, b3.x, b3.y, b3.z, b3.w};
    int4 d0 = *(const int4*)(kc + 128 + boff + 0);
    int4 d1 = *(const int4*)(kc + 128 + boff + 32);
    v8i b64 = {d0.x, d0.y, d0.z, d0.w, d1.x, d1.y, d1.z, d1.w};

    v8f oacc = {0.f, 0.f, 0.f, 0.f, 0.f, 0.f, 0.f, 0.f};
    const int aoff = hi ? 16 : 0;
    const int c0 = ct * 16;

    for (int h = 0; h < H_; ++h) {
        const unsigned char* qa = QpS + h * (16 * QK_) + n * QK_;
        int2 e0 = *(const int2*)(qa + aoff + 0);
        int2 e1 = *(const int2*)(qa + aoff + 8);
        int2 e2 = *(const int2*)(qa + aoff + 32);
        int2 e3 = *(const int2*)(qa + aoff + 40);
        int2 e4 = *(const int2*)(qa + aoff + 64);
        int2 e5 = *(const int2*)(qa + aoff + 72);
        int2 e6 = *(const int2*)(qa + aoff + 96);
        int2 e7 = *(const int2*)(qa + aoff + 104);
        v16i a128 = {e0.x, e0.y, e1.x, e1.y, e2.x, e2.y, e3.x, e3.y,
                     e4.x, e4.y, e5.x, e5.y, e6.x, e6.y, e7.x, e7.y};
        int2 f0 = *(const int2*)(qa + 128 + aoff + 0);
        int2 f1 = *(const int2*)(qa + 128 + aoff + 8);
        int2 f2 = *(const int2*)(qa + 128 + aoff + 32);
        int2 f3 = *(const int2*)(qa + 128 + aoff + 40);
        v8i a64 = {f0.x, f0.y, f1.x, f1.y, f2.x, f2.y, f3.x, f3.y};

        v8f acc = {0.f, 0.f, 0.f, 0.f, 0.f, 0.f, 0.f, 0.f};
        acc = __builtin_amdgcn_wmma_f32_16x16x128_fp8_fp8(a128, b128, (short)0, acc,
                                                          false, false);
        acc = __builtin_amdgcn_wmma_f32_16x16x64_fp8_fp8(a64, b64, (short)0, acc,
                                                         false, false);

        const float* hp = hwT + h * T_ + r0 + rowoff;
        #pragma unroll
        for (int v = 0; v < 8; ++v) {
            float val = fmaxf(acc[v], 0.0f);
            int row = r0 + v + rowoff;
            int col = c0 + n;
            if (col > row) val = 0.0f;
            oacc[v] += hp[v] * val;
        }
    }

    #pragma unroll
    for (int v = 0; v < 8; ++v)
        out[(long)(r0 + v + rowoff) * T_ + (c0 + n)] = oacc[v];
}

extern "C" void kernel_launch(void* const* d_in, const int* in_sizes, int n_in,
                              void* d_out, int out_size, void* d_ws, size_t ws_size,
                              hipStream_t stream) {
    const float* x       = (const float*)d_in[0];
    const float* wqdown  = (const float*)d_in[1];
    const float* wqup    = (const float*)d_in[2];
    const float* wqr     = (const float*)d_in[3];
    const float* wkdown  = (const float*)d_in[4];
    const float* wkup    = (const float*)d_in[5];
    const float* wkr     = (const float*)d_in[6];
    const float* wh      = (const float*)d_in[7];
    float* out = (float*)d_out;

    char* base = (char*)d_ws;
    size_t off = 0;
    auto alloc = [&](size_t bytes) -> void* {
        off = (off + 255) & ~(size_t)255;
        void* p = base + off;
        off += bytes;
        return p;
    };

    float* ql    = (float*)alloc((size_t)T_ * L_ * 4);
    float* ckv   = (float*)alloc((size_t)T_ * L_ * 4);
    float* qtmp  = (float*)alloc((size_t)T_ * D_ * 4);
    float* qrtmp = (float*)alloc((size_t)T_ * H_ * RD_ * 4);
    float* ktmp  = (float*)alloc((size_t)T_ * HD_ * 4);
    float* krtmp = (float*)alloc((size_t)T_ * RD_ * 4);
    float* hwlin = (float*)alloc((size_t)T_ * H_ * 4);
    float* hwT   = (float*)alloc((size_t)H_ * T_ * 4);
    unsigned char* Qp = (unsigned char*)alloc((size_t)H_ * T_ * QK_);
    unsigned char* Kp = (unsigned char*)alloc((size_t)T_ * QK_);
    unsigned short* xhi   = (unsigned short*)alloc((size_t)T_ * D_ * 2);
    unsigned short* xlo   = (unsigned short*)alloc((size_t)T_ * D_ * 2);
    unsigned short* qlhi  = (unsigned short*)alloc((size_t)T_ * L_ * 2);
    unsigned short* qllo  = (unsigned short*)alloc((size_t)T_ * L_ * 2);
    unsigned short* ckvhi = (unsigned short*)alloc((size_t)T_ * L_ * 2);
    unsigned short* ckvlo = (unsigned short*)alloc((size_t)T_ * L_ * 2);
    unsigned short* wqdTh = (unsigned short*)alloc((size_t)L_ * D_ * 2);
    unsigned short* wqdTl = (unsigned short*)alloc((size_t)L_ * D_ * 2);
    unsigned short* wkdTh = (unsigned short*)alloc((size_t)L_ * D_ * 2);
    unsigned short* wkdTl = (unsigned short*)alloc((size_t)L_ * D_ * 2);
    unsigned short* wkrTh = (unsigned short*)alloc((size_t)RD_ * D_ * 2);
    unsigned short* wkrTl = (unsigned short*)alloc((size_t)RD_ * D_ * 2);
    unsigned short* wquTh = (unsigned short*)alloc((size_t)D_ * L_ * 2);
    unsigned short* wquTl = (unsigned short*)alloc((size_t)D_ * L_ * 2);
    unsigned short* wqrTh = (unsigned short*)alloc((size_t)H_ * RD_ * L_ * 2);
    unsigned short* wqrTl = (unsigned short*)alloc((size_t)H_ * RD_ * L_ * 2);
    unsigned short* wkuTh = (unsigned short*)alloc((size_t)HD_ * L_ * 2);
    unsigned short* wkuTl = (unsigned short*)alloc((size_t)HD_ * L_ * 2);
    unsigned short* whTh  = (unsigned short*)alloc((size_t)H_ * L_ * 2);
    unsigned short* whTl  = (unsigned short*)alloc((size_t)H_ * L_ * 2);

    auto cdiv = [](int a, int b) { return (a + b - 1) / b; };
    auto ggrid = [&](int M, int N) { return dim3(cdiv(N, 64), M / 128); };

    split_bf16  <<<cdiv(T_ * D_, 256), 256, 0, stream>>>(x, xhi, xlo, T_ * D_);
    split_bf16_T<<<cdiv(D_ * L_, 256), 256, 0, stream>>>(wqdown, wqdTh, wqdTl, D_, L_);
    split_bf16_T<<<cdiv(D_ * L_, 256), 256, 0, stream>>>(wkdown, wkdTh, wkdTl, D_, L_);
    split_bf16_T<<<cdiv(D_ * RD_, 256), 256, 0, stream>>>(wkr, wkrTh, wkrTl, D_, RD_);

    gemm_bf16x3<4><<<ggrid(T_, L_),  256, 0, stream>>>(xhi, xlo, wqdTh, wqdTl, ql,    T_, L_,  D_);
    gemm_bf16x3<4><<<ggrid(T_, L_),  256, 0, stream>>>(xhi, xlo, wkdTh, wkdTl, ckv,   T_, L_,  D_);
    gemm_bf16x3<4><<<ggrid(T_, RD_), 256, 0, stream>>>(xhi, xlo, wkrTh, wkrTl, krtmp, T_, RD_, D_);

    split_bf16  <<<cdiv(T_ * L_, 256), 256, 0, stream>>>(ql,  qlhi,  qllo,  T_ * L_);
    split_bf16  <<<cdiv(T_ * L_, 256), 256, 0, stream>>>(ckv, ckvhi, ckvlo, T_ * L_);
    split_bf16_T<<<cdiv(L_ * D_, 256), 256, 0, stream>>>(wqup, wquTh, wquTl, L_, D_);
    split_bf16_T<<<cdiv(L_ * H_ * RD_, 256), 256, 0, stream>>>(wqr, wqrTh, wqrTl, L_, H_ * RD_);
    split_bf16_T<<<cdiv(L_ * HD_, 256), 256, 0, stream>>>(wkup, wkuTh, wkuTl, L_, HD_);
    split_bf16_T<<<cdiv(L_ * H_, 256), 256, 0, stream>>>(wh, whTh, whTl, L_, H_);

    gemm_bf16x3<4><<<ggrid(T_, D_),       256, 0, stream>>>(qlhi,  qllo,  wquTh, wquTl, qtmp,  T_, D_,       L_);
    gemm_bf16x3<4><<<ggrid(T_, H_ * RD_), 256, 0, stream>>>(qlhi,  qllo,  wqrTh, wqrTl, qrtmp, T_, H_ * RD_, L_);
    gemm_bf16x3<1><<<dim3(1, T_ / 128),   256, 0, stream>>>(qlhi,  qllo,  whTh,  whTl,  hwlin, T_, H_,       L_);
    gemm_bf16x3<4><<<ggrid(T_, HD_),      256, 0, stream>>>(ckvhi, ckvlo, wkuTh, wkuTl, ktmp,  T_, HD_,      L_);

    hw_sigmoid_t<<<cdiv(T_ * H_, 256), 256, 0, stream>>>(hwlin, hwT);

    const float s128 = 0.08838834764831845f;
    const float s64  = 0.125f;
    fwht128_quant<<<(T_ * H_) / 8, 256, 0, stream>>>(qtmp, D_,  Qp, (long)T_ * QK_, H_, s128);
    fwht128_quant<<<T_ / 8,        256, 0, stream>>>(ktmp, HD_, Kp, 0,              1,  s128);
    rope_fwht64_quant<<<(T_ * H_) / 8, 256, 0, stream>>>(qrtmp, H_ * RD_, Qp, (long)T_ * QK_, H_, s64);
    rope_fwht64_quant<<<T_ / 8,        256, 0, stream>>>(krtmp, RD_,      Kp, 0,              1,  s64);

    attn_scores<<<dim3(T_ / 128, T_ / 16), 256, 0, stream>>>(Qp, Kp, hwT, out);
}
